// VQVAE_62234076119863
// MI455X (gfx1250) — matrix operand order repaired and checked by reading, not compiled, not run on hardware
//
#include <hip/hip_runtime.h>

// ---------------------------------------------------------------------------
// CDNA5 (gfx1250) VQ-VAE forward.  Matrix work via v_wmma_f32_16x16x32_bf16;
// conv weight tiles staged into LDS with GLOBAL_LOAD_ASYNC_TO_LDS when the
// toolchain exposes the gfx1250 async builtins.
// ---------------------------------------------------------------------------

typedef __attribute__((ext_vector_type(16))) __bf16 v16bf;
typedef __attribute__((ext_vector_type(8)))  float  v8f;
typedef int v4i_vs __attribute__((vector_size(16)));   // matches builtin pointee type

__device__ __forceinline__ __bf16 f2bf(float f) { return static_cast<__bf16>(f); }

#define BAD_COORD (-0x20000000)

#if defined(__has_builtin)
#if __has_builtin(__builtin_amdgcn_global_load_async_to_lds_b128)
#define HAVE_ASYNC_LDS 1
#endif
#endif

// ---------------------------------------------------------------------------
// Implicit-GEMM conv:  GEMM M = Cout, N = NB*Ho*Wo, K = Cin*KH*KW.
// Block: 128 threads = 4 waves, all sharing one M-tile (weights staged in LDS).
// Wave tile: 16(M) x 64(N) = 4 WMMA accumulators.
// act: 0 = none, 1 = relu, 2 = sigmoid
// ---------------------------------------------------------------------------
__global__ void k_conv_wmma(const float* __restrict__ X, const float* __restrict__ W,
                            const float* __restrict__ Bias, float* __restrict__ Y,
                            int NB, int Cin, int H, int Wd,
                            int Cout, int Ho, int Wo,
                            int KH, int KW, int stride, int pad, int act) {
  __shared__ float lwf[16][36];        // 16 M-rows x 32 K f32 (144B row stride)

  const int tid  = threadIdx.x;
  const int lane = tid & 31;
  const int wave = tid >> 5;
  const int half = lane >> 4;          // 0: lanes 0-15, 1: lanes 16-31
  const int lid  = lane & 15;
  const int mtile = blockIdx.y;
  const int khw  = KH * KW;
  const int Ktot = Cin * khw;
  const int HWin = H * Wd;
  const int HWo  = Ho * Wo;
  const int Ncols = NB * HWo;
  const int colW = (blockIdx.x * 4 + wave) * 64;

  // Per-lane column descriptors for the 4 N-subtiles (column index = lid).
  bool tok[4]; int ihb[4], iwb[4], xb[4], ob[4];
#pragma unroll
  for (int t = 0; t < 4; ++t) {
    int col = colW + t * 16 + lid;
    bool ok = col < Ncols;
    int c2  = ok ? col : 0;
    int n   = c2 / HWo;
    int rem = c2 - n * HWo;
    int oh  = rem / Wo;
    int ow  = rem - oh * Wo;
    tok[t] = ok;
    ihb[t] = ok ? (oh * stride - pad) : BAD_COORD;   // poison -> bounds test fails
    iwb[t] = ow * stride - pad;
    xb[t]  = n * Cin * HWin;
    ob[t]  = (n * Cout * Ho + oh) * Wo + ow;
  }

  // Cooperative weight fill coordinates: each thread owns 4 consecutive K cols.
  const int frow = tid >> 3;                 // 0..15
  const int fcol = (tid & 7) * 4;            // 0,4,...,28
  const int fco  = mtile * 16 + frow;
  const bool fok = fco < Cout;
  const int fbase = fok ? fco * Ktot : 0;

  v8f acc0 = {}, acc1 = {}, acc2 = {}, acc3 = {};

  for (int k0 = 0; k0 < Ktot; k0 += 32) {
    const bool fullchunk = (k0 + 32) <= Ktot;        // uniform across block
    __syncthreads();                                 // previous consumption done
    if (fullchunk) {
#if defined(HAVE_ASYNC_LDS)
      // 16B async DMA: global weight row chunk -> LDS, no VGPR round trip.
      __builtin_amdgcn_global_load_async_to_lds_b128(
          (__attribute__((address_space(1))) v4i_vs*)(unsigned long long)(W + fbase + k0 + fcol),
          (__attribute__((address_space(3))) v4i_vs*)(unsigned int)(unsigned long long)(&lwf[frow][fcol]),
          0, 0);
      asm volatile("s_wait_asynccnt 0x0" ::: "memory");
#else
      const float4 wv4 = *(const float4*)(W + fbase + k0 + fcol);
      lwf[frow][fcol + 0] = wv4.x; lwf[frow][fcol + 1] = wv4.y;
      lwf[frow][fcol + 2] = wv4.z; lwf[frow][fcol + 3] = wv4.w;
#endif
    } else {
#pragma unroll
      for (int j = 0; j < 4; ++j) {
        int kk = k0 + fcol + j;
        lwf[frow][fcol + j] = (fok && kk < Ktot) ? W[fbase + kk] : 0.0f;
      }
    }
    __builtin_prefetch(W + fbase + k0 + 32 + fcol, 0, 0);
    __syncthreads();

    // ---- A fragment from LDS (one contiguous 16-K run per lane) ----------
    v16bf af;
#pragma unroll
    for (int e = 0; e < 16; ++e) {
      int kc = half * 16 + e;
      af[e] = f2bf(lwf[lid][kc]);
    }

    // ---- decompose the 16 contiguous K indices this lane supplies for B --
    int offe[16], rr[16], ss[16];
    {
      int ks  = k0 + half * 16;
      int ci  = ks / khw;
      int rem = ks - ci * khw;
      int r   = rem / KW;
      int s   = rem - r * KW;
#pragma unroll
      for (int e = 0; e < 16; ++e) {
        bool kv = (k0 + half * 16 + e) < Ktot;
        rr[e]   = kv ? r : BAD_COORD;
        ss[e]   = s;
        offe[e] = ci * HWin + r * Wd + s;
        if (++s == KW) { s = 0; if (++r == KH) { r = 0; ++ci; } }
      }
    }

    // ---- B fragments (im2col gather) + 4 WMMAs ---------------------------
#pragma unroll
    for (int t = 0; t < 4; ++t) {
      const int base = xb[t] + ihb[t] * Wd + iwb[t];
      v16bf bf;
#pragma unroll
      for (int e = 0; e < 16; ++e) {
        float xv = 0.0f;
        if ((unsigned)(ihb[t] + rr[e]) < (unsigned)H &&
            (unsigned)(iwb[t] + ss[e]) < (unsigned)Wd)
          xv = X[base + offe[e]];
        bf[e] = f2bf(xv);
      }
      if (t == 0) acc0 = __builtin_amdgcn_wmma_f32_16x16x32_bf16(false, af, false, bf, (short)0, acc0, false, false);
      if (t == 1) acc1 = __builtin_amdgcn_wmma_f32_16x16x32_bf16(false, af, false, bf, (short)0, acc1, false, false);
      if (t == 2) acc2 = __builtin_amdgcn_wmma_f32_16x16x32_bf16(false, af, false, bf, (short)0, acc2, false, false);
      if (t == 3) acc3 = __builtin_amdgcn_wmma_f32_16x16x32_bf16(false, af, false, bf, (short)0, acc3, false, false);
    }
  }

  // ---- store: D VGPR r -> row M = half*8 + r, column = lid ---------------
#pragma unroll
  for (int t = 0; t < 4; ++t) {
    if (!tok[t]) continue;
    v8f a = (t == 0) ? acc0 : (t == 1) ? acc1 : (t == 2) ? acc2 : acc3;
#pragma unroll
    for (int r = 0; r < 8; ++r) {
      int co = mtile * 16 + half * 8 + r;
      if (co < Cout) {
        float v = a[r] + Bias[co];
        if (act == 1) v = fmaxf(v, 0.0f);
        else if (act == 2) v = 1.0f / (1.0f + __expf(-v));
        Y[ob[t] + co * HWo] = v;
      }
    }
  }
}

// ---------------------------------------------------------------------------
// ConvTranspose2d (k=4, stride=2, pad=1) as 4 parity-class GEMM convs.
// blockIdx.z = parity class (py,px); per class each output sees a 2x2 tap set:
//   kh = cy + 2*th,  ih = m + py - th   (cy = (py+1)&1),  K = Cin*4.
// W layout: (Cin, Cout, 4, 4) -- strided gather, so synchronous LDS staging.
// ---------------------------------------------------------------------------
__global__ void k_convt_wmma(const float* __restrict__ X, const float* __restrict__ W,
                             const float* __restrict__ Bias, float* __restrict__ Y,
                             int NB, int Cin, int Hin, int Win, int Cout, int act) {
  __shared__ __bf16 lw[16][40];

  const int tid  = threadIdx.x;
  const int lane = tid & 31;
  const int wave = tid >> 5;
  const int half = lane >> 4;
  const int lid  = lane & 15;
  const int mtile = blockIdx.y;
  const int py = (int)blockIdx.z >> 1, px = (int)blockIdx.z & 1;
  const int cy = (py + 1) & 1, cx = (px + 1) & 1;
  const int Ktot = Cin * 4;
  const int HWin = Hin * Win;
  const int Hout = 2 * Hin, Wout = 2 * Win;
  const int Ncols = NB * HWin;
  const int colW = (blockIdx.x * 4 + wave) * 64;

  bool tok[4]; int mm[4], qq[4], base[4], ob[4];
#pragma unroll
  for (int t = 0; t < 4; ++t) {
    int col = colW + t * 16 + lid;
    bool ok = col < Ncols;
    int c2  = ok ? col : 0;
    int n   = c2 / HWin;
    int rem = c2 - n * HWin;
    int m   = rem / Win;
    int q   = rem - m * Win;
    tok[t] = ok;
    mm[t]  = ok ? m : BAD_COORD;
    qq[t]  = q;
    base[t] = n * Cin * HWin + m * Win + q;
    int ho = 2 * m + py, wo = 2 * q + px;
    ob[t]  = (n * Cout * Hout + ho) * Wout + wo;
  }

  const int frow = tid >> 3;
  const int fcol = (tid & 7) * 4;
  const int fco  = mtile * 16 + frow;
  const bool fok = fco < Cout;

  v8f acc0 = {}, acc1 = {}, acc2 = {}, acc3 = {};

  for (int k0 = 0; k0 < Ktot; k0 += 32) {
    __syncthreads();
#pragma unroll
    for (int j = 0; j < 4; ++j) {
      int kk = k0 + fcol + j;
      float wv = 0.0f;
      if (fok && kk < Ktot) {
        int ci = kk >> 2, th = (kk >> 1) & 1, tw = kk & 1;
        int kh = cy + 2 * th, kw2 = cx + 2 * tw;
        wv = W[((ci * Cout + fco) * 4 + kh) * 4 + kw2];
      }
      lw[frow][fcol + j] = f2bf(wv);
    }
    __syncthreads();

    v16bf af;
#pragma unroll
    for (int e = 0; e < 16; ++e) {
      int kc = half * 16 + e;
      af[e] = lw[lid][kc];
    }

    int offe[16], dh[16], dw[16];
#pragma unroll
    for (int e = 0; e < 16; ++e) {
      int kk = k0 + half * 16 + e;
      int ci = kk >> 2, th = (kk >> 1) & 1, tw = kk & 1;
      dh[e] = (kk < Ktot) ? (py - th) : BAD_COORD;
      dw[e] = px - tw;
      offe[e] = ci * HWin + (py - th) * Win + (px - tw);
    }

#pragma unroll
    for (int t = 0; t < 4; ++t) {
      v16bf bf;
#pragma unroll
      for (int e = 0; e < 16; ++e) {
        float xv = 0.0f;
        if ((unsigned)(mm[t] + dh[e]) < (unsigned)Hin &&
            (unsigned)(qq[t] + dw[e]) < (unsigned)Win)
          xv = X[base[t] + offe[e]];
        bf[e] = f2bf(xv);
      }
      if (t == 0) acc0 = __builtin_amdgcn_wmma_f32_16x16x32_bf16(false, af, false, bf, (short)0, acc0, false, false);
      if (t == 1) acc1 = __builtin_amdgcn_wmma_f32_16x16x32_bf16(false, af, false, bf, (short)0, acc1, false, false);
      if (t == 2) acc2 = __builtin_amdgcn_wmma_f32_16x16x32_bf16(false, af, false, bf, (short)0, acc2, false, false);
      if (t == 3) acc3 = __builtin_amdgcn_wmma_f32_16x16x32_bf16(false, af, false, bf, (short)0, acc3, false, false);
    }
  }

  const int HWout = Hout * Wout;
#pragma unroll
  for (int t = 0; t < 4; ++t) {
    if (!tok[t]) continue;
    v8f a = (t == 0) ? acc0 : (t == 1) ? acc1 : (t == 2) ? acc2 : acc3;
#pragma unroll
    for (int r = 0; r < 8; ++r) {
      int co = mtile * 16 + half * 8 + r;
      if (co < Cout) {
        float v = a[r] + Bias[co];
        if (act == 1) v = fmaxf(v, 0.0f);
        else if (act == 2) v = 1.0f / (1.0f + __expf(-v));
        Y[ob[t] + co * HWout] = v;
      }
    }
  }
}

// ---------------------------------------------------------------------------
// BatchNorm (training-mode batch stats): one block per channel.
// ---------------------------------------------------------------------------
__global__ void k_bn_stats(const float* __restrict__ X, float* __restrict__ mean,
                           float* __restrict__ rstd, int NB, int C, int HW) {
  const int c = blockIdx.x;
  float s = 0.0f, s2 = 0.0f;
  for (int n = 0; n < NB; ++n) {
    const float* xc = X + (n * C + c) * HW;
    for (int p = threadIdx.x; p < HW; p += blockDim.x) {
      float v = xc[p];
      s += v; s2 = fmaf(v, v, s2);
    }
  }
  __shared__ float bs[256], bs2[256];
  bs[threadIdx.x] = s; bs2[threadIdx.x] = s2;
  __syncthreads();
  for (int o = 128; o > 0; o >>= 1) {
    if ((int)threadIdx.x < o) { bs[threadIdx.x] += bs[threadIdx.x + o]; bs2[threadIdx.x] += bs2[threadIdx.x + o]; }
    __syncthreads();
  }
  if (threadIdx.x == 0) {
    float inv = 1.0f / (float)(NB * HW);
    float m = bs[0] * inv;
    float var = bs2[0] * inv - m * m;
    mean[c] = m;
    rstd[c] = rsqrtf(var + 1e-5f);
  }
}

// grid: (HW/256, C, NB)  -> channel from blockIdx.y (no div/mod).
__global__ void k_bn_apply(const float* __restrict__ X, const float* __restrict__ mean,
                           const float* __restrict__ rstd, const float* __restrict__ g,
                           const float* __restrict__ b, const float* __restrict__ res,
                           float* __restrict__ out, int C, int HW, int relu) {
  const int c = blockIdx.y;
  const int p = blockIdx.x * blockDim.x + threadIdx.x;
  if (p >= HW) return;
  const int i = (blockIdx.z * C + c) * HW + p;
  float v = (X[i] - mean[c]) * rstd[c] * g[c] + b[c];
  if (res) v += res[i];
  if (relu) v = fmaxf(v, 0.0f);
  out[i] = v;
}

// ---------------------------------------------------------------------------
// Vector quantize losses.  One wave per (n,h,w) row of z (D=64, K=512 codes).
// Codebook staged through LDS in 128-code (32KB) chunks.
// Forward losses: cb = cm = mse(q, e); loss = 1.25 * mse.
// ---------------------------------------------------------------------------
__global__ void k_vq(const float* __restrict__ Z, const float* __restrict__ CB,
                     float* __restrict__ acc, int NB, int C, int HW, int K) {
  __shared__ float se[8][64];
  __shared__ float cbs[128][64];
  const int wave = threadIdx.x >> 5, lane = threadIdx.x & 31;
  const int row = blockIdx.x * 8 + wave;
  const int n = row / HW, p = row - n * HW;
  float e0 = Z[(n * C + lane) * HW + p];
  float e1 = Z[(n * C + lane + 32) * HW + p];
  se[wave][lane] = e0;
  se[wave][lane + 32] = e1;

  float bestd = 3.4e38f; int bestj = 0;
  for (int c0 = 0; c0 < K; c0 += 128) {
    __syncthreads();
    const float4* src = (const float4*)(CB + c0 * 64);
    float4* dst = (float4*)&cbs[0][0];
    for (int v = threadIdx.x; v < 128 * 16; v += 256) dst[v] = src[v];
    __syncthreads();
    for (int jj = lane; jj < 128; jj += 32) {
      int j = c0 + jj;
      float d = 0.0f;
#pragma unroll 8
      for (int dd = 0; dd < 64; ++dd) {
        float t = se[wave][dd] - cbs[jj][dd];
        d = fmaf(t, t, d);
      }
      if (d < bestd || (d == bestd && j < bestj)) { bestd = d; bestj = j; }
    }
  }
#pragma unroll
  for (int off = 16; off > 0; off >>= 1) {
    float od = __shfl_xor(bestd, off, 32);
    int   oj = __shfl_xor(bestj, off, 32);
    if (od < bestd || (od == bestd && oj < bestj)) { bestd = od; bestj = oj; }
  }
  float d0 = e0 - CB[bestj * 64 + lane];
  float d1 = e1 - CB[bestj * 64 + lane + 32];
  float s = d0 * d0 + d1 * d1;
#pragma unroll
  for (int off = 16; off > 0; off >>= 1) s += __shfl_xor(s, off, 32);
  if (lane == 0) atomicAdd(acc, s);
}

__global__ void k_zero1(float* p) { if (threadIdx.x == 0) p[0] = 0.0f; }

__global__ void k_vq_final(const float* __restrict__ acc, float* __restrict__ out, float invn) {
  if (threadIdx.x == 0) {
    float mse = acc[0] * invn;
    out[0] = 1.25f * mse;  // loss = cb + 0.25*cm
    out[1] = mse;          // codebook_loss
    out[2] = mse;          // commitment_loss
  }
}

// ---------------------------------------------------------------------------
extern "C" void kernel_launch(void* const* d_in, const int* in_sizes, int n_in,
                              void* d_out, int out_size, void* d_ws, size_t ws_size,
                              hipStream_t stream) {
  const float* x      = (const float*)d_in[0];
  // encoder params
  const float* e_c1w = (const float*)d_in[1];  const float* e_c1b = (const float*)d_in[2];
  const float* e_g1  = (const float*)d_in[3];  const float* e_b1  = (const float*)d_in[4];
  const float* e_c2w = (const float*)d_in[5];  const float* e_c2b = (const float*)d_in[6];
  const float* e_g2  = (const float*)d_in[7];  const float* e_b2  = (const float*)d_in[8];
  // enc rb1: 9..14, enc rb2: 15..20
  const float* e_c3w = (const float*)d_in[21]; const float* e_c3b = (const float*)d_in[22];
  // decoder params
  const float* d_c1w = (const float*)d_in[23]; const float* d_c1b = (const float*)d_in[24];
  const float* d_g1  = (const float*)d_in[25]; const float* d_b1  = (const float*)d_in[26];
  // dec rb1: 27..32, dec rb2: 33..38
  const float* dc2w  = (const float*)d_in[39]; const float* dc2b  = (const float*)d_in[40];
  const float* d_g2  = (const float*)d_in[41]; const float* d_b2  = (const float*)d_in[42];
  const float* dc3w  = (const float*)d_in[43]; const float* dc3b  = (const float*)d_in[44];
  const float* cbk   = (const float*)d_in[45];

  float* ws  = (float*)d_ws;
  float* A   = ws;                    // 16*128*64*64 = 8,388,608 floats
  float* Bm  = ws + 8388608;          // 16*256*32*32 = 4,194,304
  float* Cm  = Bm + 4194304;          // 4,194,304
  float* st  = Cm + 4194304;          // mean[256] + rstd[256]
  float* acc = st + 512;              // loss accumulator
  float* Dm  = A;                     // alias (A free between enc conv2 and dc2)
  float* Zb  = A + 4194304;           // alias, z = (16,64,32,32) = 1,048,576

  auto conv = [&](const float* X, const float* W, const float* B, float* Y,
                  int NB, int Cin, int H, int Wd, int Cout, int Ho, int Wo,
                  int KH, int KW, int s, int p, int act) {
    int Ncols = NB * Ho * Wo;
    dim3 g((unsigned)((Ncols + 255) / 256), (unsigned)((Cout + 15) / 16), 1);
    k_conv_wmma<<<g, 128, 0, stream>>>(X, W, B, Y, NB, Cin, H, Wd, Cout, Ho, Wo, KH, KW, s, p, act);
  };
  auto convt = [&](const float* X, const float* W, const float* B, float* Y,
                   int NB, int Cin, int Hin, int Win, int Cout, int act) {
    int Ncols = NB * Hin * Win;
    dim3 g((unsigned)((Ncols + 255) / 256), (unsigned)((Cout + 15) / 16), 4);
    k_convt_wmma<<<g, 128, 0, stream>>>(X, W, B, Y, NB, Cin, Hin, Win, Cout, act);
  };
  auto bn = [&](const float* X, const float* g_, const float* b_, const float* res,
                float* out, int NB, int C, int HW, int relu) {
    k_bn_stats<<<C, 256, 0, stream>>>(X, st, st + 256, NB, C, HW);
    dim3 g((unsigned)((HW + 255) / 256), (unsigned)C, (unsigned)NB);
    k_bn_apply<<<g, 256, 0, stream>>>(X, st, st + 256, g_, b_, res, out, C, HW, relu);
  };
  auto resblock = [&](float* M, int pi) {
    const float* c1w = (const float*)d_in[pi + 0]; const float* c1b = (const float*)d_in[pi + 1];
    const float* c2w = (const float*)d_in[pi + 2]; const float* c2b = (const float*)d_in[pi + 3];
    const float* bg  = (const float*)d_in[pi + 4]; const float* bb  = (const float*)d_in[pi + 5];
    conv(M,  c1w, c1b, Cm, 16, 256, 32, 32, 256, 32, 32, 3, 3, 1, 1, 0);
    bn(Cm, bg, bb, nullptr, Cm, 16, 256, 1024, 1);
    conv(Cm, c2w, c2b, Dm, 16, 256, 32, 32, 256, 32, 32, 1, 1, 1, 0, 0);
    bn(Dm, bg, bb, M, M, 16, 256, 1024, 0);   // out = bn(conv2) + residual
  };

  // ---------------- encoder ----------------
  conv(x, e_c1w, e_c1b, A, 16, 3, 128, 128, 128, 64, 64, 4, 4, 2, 1, 0);
  bn(A, e_g1, e_b1, nullptr, A, 16, 128, 4096, 1);
  conv(A, e_c2w, e_c2b, Bm, 16, 128, 64, 64, 256, 32, 32, 4, 4, 2, 1, 0);
  bn(Bm, e_g2, e_b2, nullptr, Bm, 16, 256, 1024, 1);
  resblock(Bm, 9);
  resblock(Bm, 15);
  conv(Bm, e_c3w, e_c3b, Zb, 16, 256, 32, 32, 64, 32, 32, 3, 3, 1, 1, 0);

  // ---------------- vector quantize (losses only; z_q = z, faithful bug) --
  k_zero1<<<1, 32, 0, stream>>>(acc);
  k_vq<<<2048, 256, 0, stream>>>(Zb, cbk, acc, 16, 64, 1024, 512);
  k_vq_final<<<1, 32, 0, stream>>>(acc, (float*)d_out + 786432, 1.0f / (16384.0f * 64.0f));

  // ---------------- decoder ----------------
  conv(Zb, d_c1w, d_c1b, Bm, 16, 64, 32, 32, 256, 32, 32, 3, 3, 1, 1, 0);
  bn(Bm, d_g1, d_b1, nullptr, Bm, 16, 256, 1024, 1);
  resblock(Bm, 27);
  resblock(Bm, 33);
  convt(Bm, dc2w, dc2b, A, 16, 256, 32, 32, 128, 0);
  bn(A, d_g2, d_b2, nullptr, A, 16, 128, 4096, 1);
  convt(A, dc3w, dc3b, (float*)d_out, 16, 128, 64, 64, 3, 2);  // fused sigmoid
}
